// RuleSegmentationHead_35304631174125
// MI455X (gfx1250) — compile-verified, matrix operand order checked by reading, not run on hardware
//
#include <hip/hip_runtime.h>
#include <hip/hip_bf16.h>

// ---------------------------------------------------------------------------
// MI455X / gfx1250 implementation.
// conv_c1 (7x7, 67->32 ch, 4x512x512 px) ~= 220 GFLOP dominates; mapped to
// v_wmma_f32_16x16x32_bf16 implicit GEMM. M = out channels (16 per tile),
// N = 16 pixels along W, K = channel chunks of 32 per conv tap.
// Each wave owns TWO row-tiles so every A (weight) fragment load feeds
// 2x the WMMAs (weights are shared across the whole workgroup).
// ---------------------------------------------------------------------------

typedef __attribute__((ext_vector_type(16))) __bf16 v16bf;
typedef __attribute__((ext_vector_type(8)))  float  v8f;

#define HH 512
#define WW 512

__device__ __forceinline__ unsigned short f2bf(float f) {
    unsigned u = __float_as_uint(f);
    u += 0x7FFFu + ((u >> 16) & 1u);          // round-to-nearest-even
    return (unsigned short)(u >> 16);
}

union Frag {                                   // one 16x32 bf16 WMMA operand
    v16bf v;
    uint4 q[2];
};
union Pack8 {
    uint4 q;
    unsigned short u[8];
};

// ---------------------------------------------------------------------------
// K0: repack conv weights into per-lane WMMA A-fragment order (bf16).
// A-matrix 16x32 bf16 layout: lane L -> row M = L&15; half i holds
// kLocal = (L>=16)*8 + (i&7) + (i>=8)*16.   (ISA 05_wmma §7.12.2)
// w1p index = (((tap*3 + chunk)*2 + mtile)*32 + lane)*16 + i
// w2p index = (tap*32 + lane)*16 + i          (out ch padded 2 -> 16)
// ---------------------------------------------------------------------------
__global__ void pack_weights(const float* __restrict__ w_c1,
                             const float* __restrict__ w_c2,
                             unsigned short* __restrict__ w1p,
                             unsigned short* __restrict__ w2p) {
    const int W1E = 49 * 3 * 2 * 32 * 16;   // 150528
    const int W2E = 49 * 32 * 16;           // 25088
    int i = blockIdx.x * 256 + threadIdx.x;
    if (i < W1E) {
        int i16 = i & 15;
        int r = i >> 4;
        int lane = r & 31; r >>= 5;
        int m = r & 1;     r >>= 1;
        int q = r % 3;
        int t = r / 3;
        int kl = ((lane >> 4) << 3) + (i16 & 7) + ((i16 >> 3) << 4);
        int c = q * 32 + kl;                 // K = channel within 96-pad
        int o = m * 16 + (lane & 15);        // out channel
        int ky = t / 7, kx = t - (t / 7) * 7;
        float v = (c < 67) ? w_c1[((o * 67 + c) * 7 + ky) * 7 + kx] : 0.0f;
        w1p[i] = f2bf(v);
    } else if (i < W1E + W2E) {
        int e = i - W1E;
        int i16 = e & 15;
        int r = e >> 4;
        int lane = r & 31;
        int t = r >> 5;
        int kl = ((lane >> 4) << 3) + (i16 & 7) + ((i16 >> 3) << 4);
        int o = lane & 15;
        int ky = t / 7, kx = t - (t / 7) * 7;
        float v = (o < 2) ? w_c2[((o * 32 + kl) * 7 + ky) * 7 + kx] : 0.0f;
        w2p[e] = f2bf(v);
    }
}

// ---------------------------------------------------------------------------
// K1: 3x3 conv 64->2 + ReLU + threshold categorization + rules gather.
// 16x16 pixel tile per block, LDS-staged in 16-channel chunks.
// ---------------------------------------------------------------------------
__global__ void aux_cat(const float* __restrict__ x_aux,
                        const float* __restrict__ w_aux,
                        const float* __restrict__ b_aux,
                        const float* __restrict__ thr0,
                        const float* __restrict__ thr1,
                        const int*   __restrict__ rules,
                        float* __restrict__ out_proc,
                        float* __restrict__ out_rc,
                        int*   __restrict__ ws_rc) {
    __shared__ float tile[16][18][18];
    int tx = threadIdx.x, ty = threadIdx.y;
    int tid = ty * 16 + tx;
    int w0 = blockIdx.x * 16, h0 = blockIdx.y * 16;
    int n = blockIdx.z;
    float acc0 = 0.f, acc1 = 0.f;
    for (int cc = 0; cc < 64; cc += 16) {
        __syncthreads();
        for (int idx = tid; idx < 16 * 324; idx += 256) {
            int c = idx / 324;
            int p = idx - c * 324;
            int py = p / 18, px = p - py * 18;
            int y = h0 + py - 1, x = w0 + px - 1;
            float v = 0.f;
            if (y >= 0 && y < HH && x >= 0 && x < WW)
                v = x_aux[(((size_t)n * 64 + cc + c) * HH + y) * WW + x];
            tile[c][py][px] = v;
        }
        __syncthreads();
        for (int c = 0; c < 16; ++c) {
            #pragma unroll
            for (int ky = 0; ky < 3; ++ky)
            #pragma unroll
            for (int kx = 0; kx < 3; ++kx) {
                float v = tile[c][ty + ky][tx + kx];
                acc0 += v * w_aux[((0 * 64 + cc + c) * 3 + ky) * 3 + kx]; // SMEM-uniform
                acc1 += v * w_aux[((1 * 64 + cc + c) * 3 + ky) * 3 + kx];
            }
        }
    }
    float p0 = fmaxf(acc0 + b_aux[0], 0.f);
    float p1 = fmaxf(acc1 + b_aux[1], 0.f);
    int cat0 = (int)(p0 >= thr0[0]) + (int)(p0 >= thr0[1]) +
               (int)(p0 >= thr0[2]) + (int)(p0 >= thr0[3]);
    int cat1 = (int)(p1 >= thr1[0]) + (int)(p1 >= thr1[1]) + (int)(p1 >= thr1[2]);
    int rc = rules[cat1 + cat0 * 4];         // n_cat1 = 4
    int y = h0 + ty, x = w0 + tx;
    size_t pix = ((size_t)n * HH + y) * WW + x;
    out_proc[((size_t)n * 2 + 0) * HH * WW + (size_t)y * WW + x] = p0;
    out_proc[((size_t)n * 2 + 1) * HH * WW + (size_t)y * WW + x] = p1;
    out_rc[pix] = (float)rc;
    ws_rc[pix] = rc;
}

// ---------------------------------------------------------------------------
// K2: conv_c1 via WMMA. Block = 4 waves covering an 8-row x 16-col pixel
// tile; wave w owns rows w and w+4 (two B tiles). Per K-step: one pair of
// A fragments (out-ch 0-15 / 16-31) feeds 4 WMMAs. K = 49 taps x 96 ch
// (67 real + zero pad) -> 147 K-steps x 4 wmma per wave.
// Output (bias folded) stored bf16 pixel-major [N][H][W][32] for conv_c2.
// ---------------------------------------------------------------------------
__global__ void __launch_bounds__(128)
conv_c1_wmma(const float* __restrict__ x_corr,
             const int* __restrict__ ws_rc,
             const float* __restrict__ act_encoding,
             const unsigned short* __restrict__ w1p,
             const float* __restrict__ b_c1,
             unsigned short* __restrict__ c1out) {
    __shared__ __align__(16) unsigned short lds[14 * 22 * 96];  // 57.75 KB
    int tid = threadIdx.x;
    int n = blockIdx.z;
    int w0 = blockIdx.x * 16, h0 = blockIdx.y * 8;

    // stage channels 0..63 (fp32 -> bf16)
    for (int idx = tid; idx < 308 * 64; idx += 128) {
        int pos = idx >> 6;
        int c = idx & 63;
        int py = pos / 22, px = pos - py * 22;
        int y = h0 + py - 3, x = w0 + px - 3;
        float v = 0.f;
        if (y >= 0 && y < HH && x >= 0 && x < WW)
            v = x_corr[(((size_t)n * 64 + c) * HH + y) * WW + x];
        lds[pos * 96 + c] = f2bf(v);
    }
    // stage rule_act channels 64..66, zero pad 67..95
    for (int pos = tid; pos < 308; pos += 128) {
        int py = pos / 22, px = pos - py * 22;
        int y = h0 + py - 3, x = w0 + px - 3;
        float a0 = 0.f, a1 = 0.f, a2 = 0.f;
        if (y >= 0 && y < HH && x >= 0 && x < WW) {
            int rc = ws_rc[((size_t)n * HH + y) * WW + x];
            a0 = act_encoding[rc * 3 + 0];
            a1 = act_encoding[rc * 3 + 1];
            a2 = act_encoding[rc * 3 + 2];
        }
        lds[pos * 96 + 64] = f2bf(a0);
        lds[pos * 96 + 65] = f2bf(a1);
        lds[pos * 96 + 66] = f2bf(a2);
        for (int j = 67; j < 96; ++j) lds[pos * 96 + j] = 0;
    }
    __syncthreads();

    int lane = tid & 31;
    int wv   = tid >> 5;                 // wave id 0..3
    int r0   = wv;                       // first owned row
    int r1   = wv + 4;                   // second owned row
    int col  = lane & 15;                // pixel column (N index)
    int bsel = (lane >> 4) << 3;         // K-run base: 0 or 8 halves

    v8f accA0 = {};                      // row r0, out-ch 0-15
    v8f accA1 = {};                      // row r0, out-ch 16-31
    v8f accB0 = {};                      // row r1, out-ch 0-15
    v8f accB1 = {};                      // row r1, out-ch 16-31
    int t = 0;
    for (int ky = 0; ky < 7; ++ky) {
        for (int kx = 0; kx < 7; ++kx, ++t) {
            int pbA = ((r0 + ky) * 22 + (col + kx)) * 96;
            int pbB = ((r1 + ky) * 22 + (col + kx)) * 96;
            // prefetch next tap's weight block (global_prefetch_b8)
            __builtin_prefetch(w1p + (size_t)((t + 1) * 3 * 2 * 32) * 16, 0, 1);
            #pragma unroll
            for (int q = 0; q < 3; ++q) {
                Frag a0, a1;             // shared A fragments (L2-resident)
                const uint4* wp0 = (const uint4*)(w1p + (size_t)(((t * 3 + q) * 2 + 0) * 32 + lane) * 16);
                const uint4* wp1 = (const uint4*)(w1p + (size_t)(((t * 3 + q) * 2 + 1) * 32 + lane) * 16);
                a0.q[0] = wp0[0]; a0.q[1] = wp0[1];
                a1.q[0] = wp1[0]; a1.q[1] = wp1[1];
                Frag bA, bB;             // B fragments: ds_load_b128 x2 each
                bA.q[0] = *(const uint4*)&lds[pbA + q * 32 + bsel];
                bA.q[1] = *(const uint4*)&lds[pbA + q * 32 + bsel + 16];
                bB.q[0] = *(const uint4*)&lds[pbB + q * 32 + bsel];
                bB.q[1] = *(const uint4*)&lds[pbB + q * 32 + bsel + 16];
                accA0 = __builtin_amdgcn_wmma_f32_16x16x32_bf16(false, a0.v, false, bA.v,
                                                                (short)0, accA0, false, false);
                accA1 = __builtin_amdgcn_wmma_f32_16x16x32_bf16(false, a1.v, false, bA.v,
                                                                (short)0, accA1, false, false);
                accB0 = __builtin_amdgcn_wmma_f32_16x16x32_bf16(false, a0.v, false, bB.v,
                                                                (short)0, accB0, false, false);
                accB1 = __builtin_amdgcn_wmma_f32_16x16x32_bf16(false, a1.v, false, bB.v,
                                                                (short)0, accB1, false, false);
            }
        }
    }
    // D layout: lanes 0-15 rows M=0..7 (VGPR j = row j); lanes 16-31 rows 8..15
    int chBase = (lane < 16) ? 0 : 8;
    int x = w0 + col;
    {
        int y = h0 + r0;
        unsigned short* dst = c1out + (((size_t)n * HH + y) * WW + x) * 32;
        Pack8 s0, s1;
        #pragma unroll
        for (int j = 0; j < 8; ++j) {
            s0.u[j] = f2bf(accA0[j] + b_c1[chBase + j]);
            s1.u[j] = f2bf(accA1[j] + b_c1[16 + chBase + j]);
        }
        *(uint4*)(dst + chBase)      = s0.q;
        *(uint4*)(dst + 16 + chBase) = s1.q;
    }
    {
        int y = h0 + r1;
        unsigned short* dst = c1out + (((size_t)n * HH + y) * WW + x) * 32;
        Pack8 s0, s1;
        #pragma unroll
        for (int j = 0; j < 8; ++j) {
            s0.u[j] = f2bf(accB0[j] + b_c1[chBase + j]);
            s1.u[j] = f2bf(accB1[j] + b_c1[16 + chBase + j]);
        }
        *(uint4*)(dst + chBase)      = s0.q;
        *(uint4*)(dst + 16 + chBase) = s1.q;
    }
}

// ---------------------------------------------------------------------------
// K3: conv_c2 via WMMA (out channels zero-padded to M=16) + flat-correction
// completion + residual add. 49 K-steps of 32 channels; wave owns 2 rows so
// each A fragment load feeds 2 WMMAs.
// ---------------------------------------------------------------------------
__global__ void __launch_bounds__(128)
conv_c2_wmma(const unsigned short* __restrict__ c1in,
             const unsigned short* __restrict__ w2p,
             const float* __restrict__ b_c2,
             const int* __restrict__ ws_rc,
             const float* __restrict__ act_encoding,
             float* __restrict__ out_final,
             float* __restrict__ out_corr) {
    __shared__ __align__(16) unsigned short lds[14 * 22 * 32];  // 19.25 KB
    int tid = threadIdx.x;
    int n = blockIdx.z;
    int w0 = blockIdx.x * 16, h0 = blockIdx.y * 8;
    // stage: each position is 32 bf16 = 64B contiguous -> pure b128 traffic
    for (int idx = tid; idx < 308 * 4; idx += 128) {
        int pos = idx >> 2;
        int part = idx & 3;
        int py = pos / 22, px = pos - py * 22;
        int y = h0 + py - 3, x = w0 + px - 3;
        uint4 v = make_uint4(0u, 0u, 0u, 0u);
        if (y >= 0 && y < HH && x >= 0 && x < WW)
            v = ((const uint4*)(c1in + (((size_t)n * HH + y) * WW + x) * 32))[part];
        ((uint4*)lds)[pos * 4 + part] = v;
    }
    __syncthreads();

    int lane = tid & 31;
    int wv   = tid >> 5;
    int r0   = wv, r1 = wv + 4;
    int col  = lane & 15;
    int bsel = (lane >> 4) << 3;

    v8f accA = {};
    v8f accB = {};
    int t = 0;
    for (int ky = 0; ky < 7; ++ky) {
        for (int kx = 0; kx < 7; ++kx, ++t) {
            int pbA = ((r0 + ky) * 22 + (col + kx)) * 32;
            int pbB = ((r1 + ky) * 22 + (col + kx)) * 32;
            Frag a;
            const uint4* wp = (const uint4*)(w2p + (size_t)(t * 32 + lane) * 16);
            a.q[0] = wp[0]; a.q[1] = wp[1];
            Frag bA, bB;
            bA.q[0] = *(const uint4*)&lds[pbA + bsel];
            bA.q[1] = *(const uint4*)&lds[pbA + bsel + 16];
            bB.q[0] = *(const uint4*)&lds[pbB + bsel];
            bB.q[1] = *(const uint4*)&lds[pbB + bsel + 16];
            accA = __builtin_amdgcn_wmma_f32_16x16x32_bf16(false, a.v, false, bA.v,
                                                           (short)0, accA, false, false);
            accB = __builtin_amdgcn_wmma_f32_16x16x32_bf16(false, a.v, false, bB.v,
                                                           (short)0, accB, false, false);
        }
    }
    // rows M=0,1 (= corr ch0, ch1) live in acc[0], acc[1] of lanes 0-15
    if (lane < 16) {
        const size_t CH = (size_t)HH * WW;
        #pragma unroll
        for (int half = 0; half < 2; ++half) {
            int y = h0 + (half ? r1 : r0);
            int x = w0 + col;
            float c0  = (half ? accB[0] : accA[0]) + b_c2[0];
            float c1v = (half ? accB[1] : accA[1]) + b_c2[1];
            float c2v = -(c0 + c1v);                   // remainder channel
            size_t pix = ((size_t)n * HH + y) * WW + x;
            int rc = ws_rc[pix];
            float a0 = act_encoding[rc * 3 + 0];
            float a1 = act_encoding[rc * 3 + 1];
            float a2 = act_encoding[rc * 3 + 2];
            size_t base = ((size_t)n * 3) * HH * WW + (size_t)y * WW + x;
            out_corr[base + 0 * CH] = c0;
            out_corr[base + 1 * CH] = c1v;
            out_corr[base + 2 * CH] = c2v;
            out_final[base + 0 * CH] = a0 + c0;
            out_final[base + 1 * CH] = a1 + c1v;
            out_final[base + 2 * CH] = a2 + c2v;
        }
    }
}

// ---------------------------------------------------------------------------
extern "C" void kernel_launch(void* const* d_in, const int* in_sizes, int n_in,
                              void* d_out, int out_size, void* d_ws, size_t ws_size,
                              hipStream_t stream) {
    const float* x_aux        = (const float*)d_in[0];
    const float* x_corr       = (const float*)d_in[1];
    const float* w_aux        = (const float*)d_in[2];
    const float* b_aux        = (const float*)d_in[3];
    const float* w_c1         = (const float*)d_in[4];
    const float* b_c1         = (const float*)d_in[5];
    const float* w_c2         = (const float*)d_in[6];
    const float* b_c2         = (const float*)d_in[7];
    const float* act_encoding = (const float*)d_in[8];
    const float* thr0         = (const float*)d_in[9];
    const float* thr1         = (const float*)d_in[10];
    const int*   rules        = (const int*)d_in[11];

    float* out = (float*)d_out;
    const size_t NP = (size_t)4 * HH * WW;   // 1,048,576 pixels
    float* out_final = out;                  // [4,3,H,W]
    float* out_rc    = out + 3 * NP;         // [4,H,W]
    float* out_corr  = out + 4 * NP;         // [4,3,H,W]
    float* out_proc  = out + 7 * NP;         // [4,2,H,W]

    char* ws = (char*)d_ws;
    int*            ws_rc = (int*)ws;                                         // 4 MB
    unsigned short* ws_c1 = (unsigned short*)(ws + (size_t)4  * 1024 * 1024); // 64 MB bf16 c1
    unsigned short* ws_w1 = (unsigned short*)(ws + (size_t)68 * 1024 * 1024); // 301056 B
    unsigned short* ws_w2 = (unsigned short*)(ws + (size_t)68 * 1024 * 1024 + 301056);

    pack_weights<<<dim3((150528 + 25088 + 255) / 256), dim3(256), 0, stream>>>(
        w_c1, w_c2, ws_w1, ws_w2);
    aux_cat<<<dim3(32, 32, 4), dim3(16, 16), 0, stream>>>(
        x_aux, w_aux, b_aux, thr0, thr1, rules, out_proc, out_rc, ws_rc);
    conv_c1_wmma<<<dim3(32, 64, 4), dim3(128), 0, stream>>>(
        x_corr, ws_rc, act_encoding, ws_w1, b_c1, ws_c1);
    conv_c2_wmma<<<dim3(32, 64, 4), dim3(128), 0, stream>>>(
        ws_c1, ws_w2, b_c2, ws_rc, act_encoding, out_final, out_corr);
}
